// SIGRegHard_15788299780871
// MI455X (gfx1250) — compile-verified, matrix operand order checked by reading, not run on hardware
//
#include <hip/hip_runtime.h>
#include <hip/hip_bf16.h>

typedef unsigned short u16;
typedef unsigned int   u32;
typedef unsigned long long u64;

typedef __attribute__((ext_vector_type(16))) __bf16 v16bf;
typedef __attribute__((ext_vector_type(8)))  float  v8f;

#define BATCH  2048
#define DIM    1024
#define POOLN  2048
#define HARDN  128
#define EXTRAN 64
#define FINALN 192
#define NT     8          // positive t offsets (t = m*0.625, m=1..8); mirror for negative half
#define DT     0.625f
#define EPSV   1e-12f

// ---------------- helpers ----------------
__device__ inline u16 f2bf(float f){
  u32 x = __float_as_uint(f);
  u32 r = x + 0x7FFFu + ((x >> 16) & 1u);     // round-to-nearest-even
  return (u16)(r >> 16);
}
__device__ inline float bf2f(u16 h){ return __uint_as_float(((u32)h) << 16); }

__device__ inline u64 splitmix64(u64 x){
  x += 0x9E3779B97F4A7C15ull;
  x = (x ^ (x >> 30)) * 0xBF58476D1CE4E5B9ull;
  x = (x ^ (x >> 27)) * 0x94D049BB133111EBull;
  return x ^ (x >> 31);
}
// counter-based standard normal (Box-Muller); structural stand-in for JAX threefry
__device__ inline float rng_normal(u64 stream, u64 idx){
  u64 r = splitmix64(splitmix64(idx ^ (stream * 0x9E3779B97F4A7C15ull)) + stream + 0x632BE59BD9B4E019ull);
  u32 a = (u32)r, b = (u32)(r >> 32);
  float u1 = ((float)a + 1.0f) * 2.3283064365386963e-10f;   // (0,1]
  float u2 = (float)b * 2.3283064365386963e-10f;            // [0,1)
  float rad = sqrtf(fmaxf(-2.0f * __logf(u1), 0.0f));
  return rad * __cosf(6.283185307179586f * u2);
}

__device__ inline float blockReduceSum(float v, float* sred){
  #pragma unroll
  for (int o = 16; o > 0; o >>= 1) v += __shfl_down(v, o, 32);
  const int lane = threadIdx.x & 31, w = threadIdx.x >> 5, nw = blockDim.x >> 5;
  if (lane == 0) sred[w] = v;
  __syncthreads();
  float r = (threadIdx.x < nw) ? sred[threadIdx.x] : 0.0f;
  if (w == 0){
    #pragma unroll
    for (int o = 16; o > 0; o >>= 1) r += __shfl_down(r, o, 32);
    if (lane == 0) sred[0] = r;
  }
  __syncthreads();
  r = sred[0];
  __syncthreads();
  return r;
}

// ---------------- K0: column means of z ----------------
__global__ void mean_kernel(const float* __restrict__ z, float* __restrict__ mean){
  int d = blockIdx.x * blockDim.x + threadIdx.x;      // 1024 threads total
  float s = 0.f;
  for (int b = 0; b < BATCH; ++b) s += z[(size_t)b * DIM + d];
  mean[d] = s * (1.0f / BATCH);
}

// ---------------- K1: zc = z - mean, split to bf16 hi/lo ----------------
__global__ void splitz_kernel(const float* __restrict__ z, const float* __restrict__ mean,
                              u16* __restrict__ zhi, u16* __restrict__ zlo){
  size_t i = (size_t)blockIdx.x * blockDim.x + threadIdx.x;
  int d = (int)(i & (DIM - 1));
  float f = z[i] - mean[d];
  u16 h = f2bf(f);
  zhi[i] = h;
  zlo[i] = f2bf(f - bf2f(h));
}

// ---------------- K2: generate + normalize A_pool (pool-major [p][d]) ----------------
__global__ void genA_kernel(const int* __restrict__ seed_ptr, float* __restrict__ Apool,
                            u16* __restrict__ Ahi, u16* __restrict__ Alo){
  __shared__ float sred[32];
  const int p = blockIdx.x;
  const u64 stream = (u64)(u32)(*seed_ptr) * 2ull + 1ull;
  float vals[4]; float sq = 0.f;
  #pragma unroll
  for (int k = 0; k < 4; ++k){
    int d = threadIdx.x + k * 256;
    float v = rng_normal(stream, (u64)p * DIM + d);
    vals[k] = v; sq += v * v;
  }
  float n = sqrtf(blockReduceSum(sq, sred));
  float inv = 1.0f / (n + EPSV);
  #pragma unroll
  for (int k = 0; k < 4; ++k){
    int d = threadIdx.x + k * 256;
    size_t idx = (size_t)p * DIM + d;
    float v = vals[k] * inv;
    Apool[idx] = v;
    u16 h = f2bf(v);
    Ahi[idx] = h;
    Alo[idx] = f2bf(v - bf2f(h));
  }
}

// ---------------- K3: fused WMMA GEMM + ECF accumulation ----------------
// Workgroup = 256 threads (8 waves). Block owns 16 pool columns; each wave covers
// BATCH/8 = 256 batch rows, processed as 8 pairs of 16-row tiles. Each LDS B fragment
// feeds 6 WMMAs (2 tiles x bf16x3 split). '#pragma unroll 1' on the pair loop keeps
// the kk loop rolled so LLVM cannot hoist LDS B-fragments across tiles (which caused
// scratch spills in the previous version).
__global__ __launch_bounds__(256) void ecf_kernel(
    const u16* __restrict__ Bhi, const u16* __restrict__ Blo,
    const u16* __restrict__ zhi, const u16* __restrict__ zlo,
    float* __restrict__ Sre, float* __restrict__ Sim){
  // 64KB LDS: [0..2047]=hi fragments, [2048..4095]=lo fragments; reused as f32 reduce scratch
  __shared__ uint4 smem[4096];
  const int tid  = threadIdx.x;
  const int lane = tid & 31;
  const int w    = tid >> 5;
  const int colbase = blockIdx.x * 16;

  // Stage B (pool columns) into LDS in fragment order: slot s -> (kstep, chunk, lane)
  for (int s = tid; s < 2048; s += 256){
    int kk = s >> 6, rem = s & 63, chunk = rem >> 5, l = rem & 31;
    int col = l & 15, khalf = l >> 4;
    size_t g = (size_t)(colbase + col) * DIM + (size_t)kk * 32 + chunk * 16 + khalf * 8;
    smem[s]        = *(const uint4*)(Bhi + g);
    smem[2048 + s] = *(const uint4*)(Blo + g);
  }
  __syncthreads();

  const int col = lane & 15, khalf = lane >> 4;
  float re[NT], im[NT];
  #pragma unroll
  for (int m = 0; m < NT; ++m){ re[m] = 0.f; im[m] = 0.f; }

  const int rowbase0 = w * (BATCH / 8);
  #pragma unroll 1
  for (int tp = 0; tp < (BATCH / 8) / 32; ++tp){      // 8 groups of 2 tiles
    const int row0 = rowbase0 + tp * 32 + col;        // A-frag: M = lane%16
    const u16* a0h = zhi + (size_t)row0 * DIM + khalf * 8;
    const u16* a0l = zlo + (size_t)row0 * DIM + khalf * 8;
    const u16* a1h = a0h + (size_t)16 * DIM;
    const u16* a1l = a0l + (size_t)16 * DIM;
    v8f acc0 = {}, acc1 = {};
    #pragma unroll 4
    for (int kk = 0; kk < DIM / 32; ++kk){
      union { uint4 q[2]; v16bf v; } bh, bl, ah0, al0, ah1, al1;
      bh.q[0]  = smem[(kk * 2 + 0) * 32 + lane];
      bh.q[1]  = smem[(kk * 2 + 1) * 32 + lane];
      bl.q[0]  = smem[2048 + (kk * 2 + 0) * 32 + lane];
      bl.q[1]  = smem[2048 + (kk * 2 + 1) * 32 + lane];
      ah0.q[0] = *(const uint4*)(a0h + kk * 32);
      ah0.q[1] = *(const uint4*)(a0h + kk * 32 + 16);
      al0.q[0] = *(const uint4*)(a0l + kk * 32);
      al0.q[1] = *(const uint4*)(a0l + kk * 32 + 16);
      ah1.q[0] = *(const uint4*)(a1h + kk * 32);
      ah1.q[1] = *(const uint4*)(a1h + kk * 32 + 16);
      al1.q[0] = *(const uint4*)(a1l + kk * 32);
      al1.q[1] = *(const uint4*)(a1l + kk * 32 + 16);
      // u = (Ahi+Alo)(Bhi+Blo) ~= Ahi*Bhi + Ahi*Blo + Alo*Bhi   (drop lo*lo ~ 2^-16)
      acc0 = __builtin_amdgcn_wmma_f32_16x16x32_bf16(false, ah0.v, false, bh.v, (short)0, acc0, false, false);
      acc0 = __builtin_amdgcn_wmma_f32_16x16x32_bf16(false, ah0.v, false, bl.v, (short)0, acc0, false, false);
      acc0 = __builtin_amdgcn_wmma_f32_16x16x32_bf16(false, al0.v, false, bh.v, (short)0, acc0, false, false);
      acc1 = __builtin_amdgcn_wmma_f32_16x16x32_bf16(false, ah1.v, false, bh.v, (short)0, acc1, false, false);
      acc1 = __builtin_amdgcn_wmma_f32_16x16x32_bf16(false, ah1.v, false, bl.v, (short)0, acc1, false, false);
      acc1 = __builtin_amdgcn_wmma_f32_16x16x32_bf16(false, al1.v, false, bh.v, (short)0, acc1, false, false);
    }
    #pragma unroll
    for (int r = 0; r < 8; ++r){
      {
        float u  = acc0[r];
        float du = DT * u;
        float cd = __cosf(du), sd = __sinf(du);
        float cr = 1.f, ci = 0.f;
        #pragma unroll
        for (int m = 0; m < NT; ++m){
          float nc = cr * cd - ci * sd;
          ci = ci * cd + cr * sd;
          cr = nc;
          re[m] += cr; im[m] += ci;
        }
      }
      {
        float u  = acc1[r];
        float du = DT * u;
        float cd = __cosf(du), sd = __sinf(du);
        float cr = 1.f, ci = 0.f;
        #pragma unroll
        for (int m = 0; m < NT; ++m){
          float nc = cr * cd - ci * sd;
          ci = ci * cd + cr * sd;
          cr = nc;
          re[m] += cr; im[m] += ci;
        }
      }
    }
  }
  // lanes l and l^16 hold complementary row halves of the same column
  #pragma unroll
  for (int m = 0; m < NT; ++m){
    re[m] += __shfl_xor(re[m], 16, 32);
    im[m] += __shfl_xor(im[m], 16, 32);
  }
  __syncthreads();                      // all waves done with B fragments -> reuse LDS
  float* redf = (float*)smem;           // [wave][16 (8 re + 8 im)][16 cols]
  if (lane < 16){
    #pragma unroll
    for (int m = 0; m < NT; ++m){
      redf[(w * 16 + m)     * 16 + lane] = re[m];
      redf[(w * 16 + 8 + m) * 16 + lane] = im[m];
    }
  }
  __syncthreads();
  {
    int c = tid & 15, m = tid >> 4;     // 256 threads: (m 0..15, c 0..15)
    float s = 0.f;
    #pragma unroll
    for (int ww = 0; ww < 8; ++ww) s += redf[(ww * 16 + m) * 16 + c];
    int gc = colbase + c;
    if (m < 8) Sre[(size_t)gc * 8 + m]       = s;
    else       Sim[(size_t)gc * 8 + (m - 8)] = s;
  }
}

// ---------------- K4: per-column loss from ECF sums ----------------
__device__ inline float loss_from_S(const float* Sre, const float* Sim, int p){
  float acc = 0.f;
  #pragma unroll
  for (int m = 1; m <= 8; ++m){
    float t   = DT * m;
    float phi = __expf(-0.5f * t * t);
    float reV = Sre[(size_t)p * 8 + (m - 1)] * (1.0f / BATCH);
    float imV = Sim[(size_t)p * 8 + (m - 1)] * (1.0f / BATCH);
    float dre = reV - phi;
    float I   = (dre * dre + imV * imV) * phi;
    acc += (m == 8) ? I : 2.0f * I;     // symmetry: I(t)=I(-t); I(0)=0; ends get weight 1/2 each
  }
  return DT * acc;
}

__global__ void loss_kernel(const float* __restrict__ Sre, const float* __restrict__ Sim,
                            float* __restrict__ losses, int ncols){
  int p = blockIdx.x * blockDim.x + threadIdx.x;
  if (p >= ncols) return;
  losses[p] = loss_from_S(Sre, Sim, p);
}

// ---------------- K5: bitonic argsort (descending loss) ----------------
__global__ __launch_bounds__(1024) void sort_kernel(const float* __restrict__ losses,
                                                    int* __restrict__ ranked){
  __shared__ float key[2048];
  __shared__ int   val[2048];
  const int tid = threadIdx.x;
  for (int i = tid; i < 2048; i += 1024){ key[i] = -losses[i]; val[i] = i; }
  __syncthreads();
  for (int k = 2; k <= 2048; k <<= 1){
    for (int j = k >> 1; j > 0; j >>= 1){
      for (int i = tid; i < 2048; i += 1024){
        int ixj = i ^ j;
        if (ixj > i){
          bool up = ((i & k) == 0);
          float a = key[i], b = key[ixj];
          if ((a > b) == up){
            key[i] = b; key[ixj] = a;
            int t = val[i]; val[i] = val[ixj]; val[ixj] = t;
          }
        }
      }
      __syncthreads();
    }
  }
  for (int i = tid; i < 2048; i += 1024) ranked[i] = val[i];
}

// ---------------- K6: classical Gram-Schmidt on top-128 columns ----------------
__global__ __launch_bounds__(1024) void gs_kernel(const float* __restrict__ Apool,
                                                  const int* __restrict__ ranked,
                                                  float* __restrict__ Q){
  __shared__ float vsh[DIM];
  __shared__ float csh[HARDN];
  const int t = threadIdx.x;            // one thread per dimension
  for (int j = 0; j < HARDN; ++j){
    float v0 = Apool[(size_t)ranked[j] * DIM + t];
    vsh[t] = v0;
    __syncthreads();
    if (t < j){                          // c_i = Q_i . v0  (classical GS: original v)
      float c = 0.f;
      for (int d = 0; d < DIM; ++d) c += Q[(size_t)t * DIM + d] * vsh[d];
      csh[t] = c;
    }
    __syncthreads();
    float proj = 0.f;
    for (int i = 0; i < j; ++i) proj += csh[i] * Q[(size_t)i * DIM + t];
    float v = v0 - proj;
    __syncthreads();
    vsh[t] = v * v;
    __syncthreads();
    for (int s = 512; s > 0; s >>= 1){ if (t < s) vsh[t] += vsh[t + s]; __syncthreads(); }
    float n = sqrtf(vsh[0]);
    __syncthreads();
    Q[(size_t)j * DIM + t] = v / fmaxf(n, EPSV);
    __syncthreads();
  }
}

// ---------------- K7: generate + normalize R ----------------
__global__ void genR_kernel(const int* __restrict__ seed_ptr, float* __restrict__ Rm){
  __shared__ float sred[32];
  const int p = blockIdx.x;
  const u64 stream = (u64)(u32)(*seed_ptr + 9991);
  float vals[4]; float sq = 0.f;
  #pragma unroll
  for (int k = 0; k < 4; ++k){
    int d = threadIdx.x + k * 256;
    float v = rng_normal(stream, (u64)p * DIM + d);
    vals[k] = v; sq += v * v;
  }
  float n = sqrtf(blockReduceSum(sq, sred));
  float inv = 1.0f / (n + EPSV);
  #pragma unroll
  for (int k = 0; k < 4; ++k){
    int d = threadIdx.x + k * 256;
    Rm[(size_t)p * DIM + d] = vals[k] * inv;
  }
}

// ---------------- K8: project R off hard, renormalize ----------------
__global__ __launch_bounds__(1024) void projR_kernel(const float* __restrict__ Q,
                                                     float* __restrict__ Rm){
  __shared__ float sred[32];
  const int r = blockIdx.x, t = threadIdx.x;
  float v0 = Rm[(size_t)r * DIM + t];
  float proj = 0.f;
  for (int i = 0; i < HARDN; ++i){
    float qi = Q[(size_t)i * DIM + t];
    float c  = blockReduceSum(qi * v0, sred);
    proj += c * qi;
  }
  float v = v0 - proj;
  float n = sqrtf(blockReduceSum(v * v, sred));
  Rm[(size_t)r * DIM + t] = v / fmaxf(n, EPSV);
}

// ---------------- K9: A_final -> bf16 hi/lo ----------------
__global__ void convfinal_kernel(const float* __restrict__ Q, const float* __restrict__ Rm,
                                 u16* __restrict__ Fhi, u16* __restrict__ Flo){
  size_t i = (size_t)blockIdx.x * blockDim.x + threadIdx.x;   // FINALN*DIM
  int p = (int)(i / DIM), d = (int)(i & (DIM - 1));
  float f = (p < HARDN) ? Q[(size_t)p * DIM + d] : Rm[(size_t)(p - HARDN) * DIM + d];
  u16 h = f2bf(f);
  Fhi[i] = h;
  Flo[i] = f2bf(f - bf2f(h));
}

// ---------------- K10: final losses + mean * B ----------------
__global__ void final_kernel(const float* __restrict__ Sre, const float* __restrict__ Sim,
                             float* __restrict__ out){
  __shared__ float sred[32];
  const int p = threadIdx.x;            // 256 threads
  float L = (p < FINALN) ? loss_from_S(Sre, Sim, p) : 0.f;
  float s = blockReduceSum(L, sred);
  if (p == 0) out[0] = s * ((float)BATCH / (float)FINALN);
}

// ---------------- launcher ----------------
extern "C" void kernel_launch(void* const* d_in, const int* in_sizes, int n_in,
                              void* d_out, int out_size, void* d_ws, size_t ws_size,
                              hipStream_t stream){
  (void)in_sizes; (void)n_in; (void)out_size; (void)ws_size;
  const float* z        = (const float*)d_in[0];
  const int*   seed_ptr = (const int*)d_in[1];
  float* out = (float*)d_out;

  char* ws = (char*)d_ws;
  size_t off = 0;
  auto alloc = [&](size_t bytes) -> void* {
    void* p = ws + off;
    off = (off + bytes + 255) & ~(size_t)255;
    return p;
  };
  float* mean  = (float*)alloc((size_t)DIM * 4);
  u16*   zhi   = (u16*)  alloc((size_t)BATCH * DIM * 2);
  u16*   zlo   = (u16*)  alloc((size_t)BATCH * DIM * 2);
  float* Apool = (float*)alloc((size_t)POOLN * DIM * 4);
  u16*   Ahi   = (u16*)  alloc((size_t)POOLN * DIM * 2);
  u16*   Alo   = (u16*)  alloc((size_t)POOLN * DIM * 2);
  float* S1re  = (float*)alloc((size_t)POOLN * 8 * 4);
  float* S1im  = (float*)alloc((size_t)POOLN * 8 * 4);
  float* loss1 = (float*)alloc((size_t)POOLN * 4);
  int*   rankd = (int*)  alloc((size_t)POOLN * 4);
  float* Q     = (float*)alloc((size_t)HARDN * DIM * 4);
  float* Rm    = (float*)alloc((size_t)EXTRAN * DIM * 4);
  u16*   Fhi   = (u16*)  alloc((size_t)FINALN * DIM * 2);
  u16*   Flo   = (u16*)  alloc((size_t)FINALN * DIM * 2);
  float* S2re  = (float*)alloc((size_t)FINALN * 8 * 4);
  float* S2im  = (float*)alloc((size_t)FINALN * 8 * 4);

  mean_kernel  <<<DIM / 256, 256, 0, stream>>>(z, mean);
  splitz_kernel<<<(BATCH * DIM) / 256, 256, 0, stream>>>(z, mean, zhi, zlo);
  genA_kernel  <<<POOLN, 256, 0, stream>>>(seed_ptr, Apool, Ahi, Alo);
  ecf_kernel   <<<POOLN / 16, 256, 0, stream>>>(Ahi, Alo, zhi, zlo, S1re, S1im);
  loss_kernel  <<<POOLN / 256, 256, 0, stream>>>(S1re, S1im, loss1, POOLN);
  sort_kernel  <<<1, 1024, 0, stream>>>(loss1, rankd);
  gs_kernel    <<<1, 1024, 0, stream>>>(Apool, rankd, Q);
  genR_kernel  <<<EXTRAN, 256, 0, stream>>>(seed_ptr, Rm);
  projR_kernel <<<EXTRAN, 1024, 0, stream>>>(Q, Rm);
  convfinal_kernel<<<(FINALN * DIM) / 256, 256, 0, stream>>>(Q, Rm, Fhi, Flo);
  ecf_kernel   <<<FINALN / 16, 256, 0, stream>>>(Fhi, Flo, zhi, zlo, S2re, S2im);
  final_kernel <<<1, 256, 0, stream>>>(S2re, S2im, out);
}
